// TriangleMultiplicativeModuleLocal_57501022159282
// MI455X (gfx1250) — compile-verified, matrix operand order checked by reading, not run on hardware
//
#include <hip/hip_runtime.h>

typedef _Float16 f16;
typedef __attribute__((ext_vector_type(16))) _Float16 v16h;
typedef __attribute__((ext_vector_type(8)))  float    v8f;

#define LL    512
#define DIM_  64
#define HID_  128
#define WIN_  4
#define NROWS ((long)LL * (long)LL)

__device__ __forceinline__ float sigf(float x) { return 1.f / (1.f + __expf(-x)); }

// ---------------------------------------------------------------------------
// Prep: build f16 N-major (transposed) weights so WMMA B-fragments are two
// contiguous b128 loads per lane; also concat the 5 projection biases.
//   W5T[n][k] : n = m*128+h over {Wl,Wr,Wlg,Wrg,Wog}, k over DIM
//   WoT[n][k] : n over DIM, k over HID
// ---------------------------------------------------------------------------
__global__ void __launch_bounds__(256)
tm_prep_kernel(const float* __restrict__ Wl, const float* __restrict__ Wr,
               const float* __restrict__ Wlg, const float* __restrict__ Wrg,
               const float* __restrict__ Wog, const float* __restrict__ Wo,
               const float* __restrict__ bl, const float* __restrict__ br,
               const float* __restrict__ blg, const float* __restrict__ brg,
               const float* __restrict__ bog,
               f16* __restrict__ W5T, f16* __restrict__ WoT, float* __restrict__ b5)
{
    int t = blockIdx.x * 256 + threadIdx.x;
    if (t < 640 * 64) {
        int n = t / 64, k = t % 64;
        int m = n / 128, h = n % 128;
        const float* W = (m == 0) ? Wl : (m == 1) ? Wr : (m == 2) ? Wlg : (m == 3) ? Wrg : Wog;
        W5T[n * 64 + k] = (f16)W[k * HID_ + h];
    }
    if (t < 64 * 128) {
        int n = t / 128, k = t % 128;
        WoT[n * 128 + k] = (f16)Wo[k * DIM_ + n];
    }
    if (t < 640) {
        int m = t / 128, h = t % 128;
        const float* bp = (m == 0) ? bl : (m == 1) ? br : (m == 2) ? blg : (m == 3) ? brg : bog;
        b5[t] = bp[h];
    }
}

// ---------------------------------------------------------------------------
// Kernel A: per block = 16 flattened (i,j) rows.
//   LN(64) -> f16 tile in LDS -> 8 waves each own one 16-wide h-chunk and run
//   the same A-fragments against 5 weight tiles (10 WMMAs/wave), then fuse
//   mask * sigmoid gating and store left_g/right_g/og as f16.
// ---------------------------------------------------------------------------
__global__ void __launch_bounds__(256)
tm_proj_kernel(const float* __restrict__ x, const int* __restrict__ smask,
               const float* __restrict__ nscale, const float* __restrict__ nbias,
               const f16* __restrict__ W5T, const float* __restrict__ b5,
               f16* __restrict__ leftg, f16* __restrict__ rightg, f16* __restrict__ og)
{
    __shared__ _Float16 xn[16 * 64];

    const int  tid     = threadIdx.x;
    const long rowbase = (long)blockIdx.x * 16;

    // ---- LayerNorm over DIM=64: 16 threads/row, 4 floats each
    {
        int  r   = tid >> 4;
        int  c   = tid & 15;
        long row = rowbase + r;
        float4 v = *(const float4*)(x + row * DIM_ + c * 4);
        float s  = v.x + v.y + v.z + v.w;
        float s2 = v.x * v.x + v.y * v.y + v.z * v.z + v.w * v.w;
        #pragma unroll
        for (int m = 1; m < 16; m <<= 1) {
            s  += __shfl_xor(s,  m, 16);
            s2 += __shfl_xor(s2, m, 16);
        }
        float mean = s * (1.f / 64.f);
        float var  = s2 * (1.f / 64.f) - mean * mean;
        float rs   = rsqrtf(var + 1e-6f);
        int d = c * 4;
        _Float16* dst = &xn[r * 64 + d];
        dst[0] = (f16)(((v.x - mean) * rs) * nscale[d + 0] + nbias[d + 0]);
        dst[1] = (f16)(((v.y - mean) * rs) * nscale[d + 1] + nbias[d + 1]);
        dst[2] = (f16)(((v.z - mean) * rs) * nscale[d + 2] + nbias[d + 2]);
        dst[3] = (f16)(((v.w - mean) * rs) * nscale[d + 3] + nbias[d + 3]);
    }
    __syncthreads();

    const int lane = tid & 31;
    const int w    = tid >> 5;     // wave id == h-chunk (0..7)
    const int lr   = lane & 15;
    const int g    = lane >> 4;

    // A fragments (16x32 f16): per-lane = two contiguous 8-half runs
    v16h a0, a1;
    {
        const _Float16* base = &xn[lr * 64];
        int ko = 8 * g;
        ((uint4*)&a0)[0] = *(const uint4*)(base + ko);
        ((uint4*)&a0)[1] = *(const uint4*)(base + ko + 16);
        ((uint4*)&a1)[0] = *(const uint4*)(base + 32 + ko);
        ((uint4*)&a1)[1] = *(const uint4*)(base + 32 + ko + 16);
    }

    v8f acc[5];
    #pragma unroll
    for (int m = 0; m < 5; ++m) {
        int   ncol = m * 128 + w * 16 + lr;
        float bias = b5[ncol];
        v8f c;
        #pragma unroll
        for (int r = 0; r < 8; ++r) c[r] = bias;
        const f16* brow = W5T + (long)ncol * 64;
        int ko = 8 * g;
        v16h b0, b1;
        ((uint4*)&b0)[0] = *(const uint4*)(brow + ko);
        ((uint4*)&b0)[1] = *(const uint4*)(brow + ko + 16);
        ((uint4*)&b1)[0] = *(const uint4*)(brow + 32 + ko);
        ((uint4*)&b1)[1] = *(const uint4*)(brow + 32 + ko + 16);
        c = __builtin_amdgcn_wmma_f32_16x16x32_f16(false, a0, false, b0, (short)0, c, false, false);
        c = __builtin_amdgcn_wmma_f32_16x16x32_f16(false, a1, false, b1, (short)0, c, false, false);
        acc[m] = c;
    }

    // Fused epilogue: mask + sigmoid gates; f16 stores
    #pragma unroll
    for (int r = 0; r < 8; ++r) {
        long row = rowbase + r + 8 * g;
        int  i   = (int)(row >> 9);
        int  j   = (int)(row & 511);
        float mk = (float)(smask[i] * smask[j]);
        long off = row * HID_ + (long)(w * 16 + lr);
        leftg[off]  = (f16)(acc[0][r] * mk * sigf(acc[2][r]));
        rightg[off] = (f16)(acc[1][r] * mk * sigf(acc[3][r]));
        og[off]     = (f16)sigf(acc[4][r]);
    }
}

// ---------------------------------------------------------------------------
// Kernel B: per block = one i, 32 j's.
//   Phase 1: banded out[i,j,h] = sum_{k=i-4..i+4} left_g[k,i,h]*right_g[k,j,h]
//            (Lv[9][128] staged in LDS), then LN(128) via 8-lane shuffles,
//            gate with og, f16 tile to LDS.
//   Phase 2: [32x128]@WoT[64x128]^T via WMMA, 8 waves = 2 M-tiles x 4 N-tiles.
// ---------------------------------------------------------------------------
__global__ void __launch_bounds__(256)
tm_band_out_kernel(const f16* __restrict__ leftg, const f16* __restrict__ rightg,
                   const f16* __restrict__ og,
                   const float* __restrict__ onscale, const float* __restrict__ onbias,
                   const f16* __restrict__ WoT, const float* __restrict__ bo,
                   float* __restrict__ out)
{
    __shared__ float    Lv[9][HID_];
    __shared__ _Float16 vt[32 * HID_];

    const int tid = threadIdx.x;
    const int i   = blockIdx.x >> 4;
    const int jt  = blockIdx.x & 15;

    // Stage the 9 left_g band vectors for this i
    for (int idx = tid; idx < 9 * HID_; idx += 256) {
        int kk = idx / HID_, h = idx % HID_;
        int k  = i - WIN_ + kk;
        float v = 0.f;
        if (k >= 0 && k < LL) v = (float)leftg[((long)k * LL + i) * HID_ + h];
        Lv[kk][h] = v;
    }
    __syncthreads();

    const int jl = tid >> 3;          // 0..31
    const int hc = (tid & 7) * 16;    // h-chunk
    const int j  = jt * 32 + jl;

    float acc[16];
    #pragma unroll
    for (int t = 0; t < 16; ++t) acc[t] = 0.f;

    for (int kk = 0; kk < 9; ++kk) {
        int k = i - WIN_ + kk;
        if (k < 0 || k >= LL) continue;   // block-uniform
        const f16* rp = rightg + ((long)k * LL + j) * HID_ + hc;
        uint4 u0 = ((const uint4*)rp)[0];
        uint4 u1 = ((const uint4*)rp)[1];
        const _Float16* r0 = (const _Float16*)&u0;
        const _Float16* r1 = (const _Float16*)&u1;
        #pragma unroll
        for (int t = 0; t < 8; ++t) acc[t]     += Lv[kk][hc + t]     * (float)r0[t];
        #pragma unroll
        for (int t = 0; t < 8; ++t) acc[8 + t] += Lv[kk][hc + 8 + t] * (float)r1[t];
    }

    // LayerNorm over HID=128 (8 lanes share one row)
    float s = 0.f, s2 = 0.f;
    #pragma unroll
    for (int t = 0; t < 16; ++t) { s += acc[t]; s2 += acc[t] * acc[t]; }
    #pragma unroll
    for (int m = 1; m < 8; m <<= 1) {
        s  += __shfl_xor(s,  m, 8);
        s2 += __shfl_xor(s2, m, 8);
    }
    float mean = s * (1.f / 128.f);
    float var  = s2 * (1.f / 128.f) - mean * mean;
    float rs   = rsqrtf(var + 1e-6f);

    const long row = (long)i * LL + j;
    const f16* op  = og + row * HID_ + hc;
    uint4 o0 = ((const uint4*)op)[0];
    uint4 o1 = ((const uint4*)op)[1];
    const _Float16* oh0 = (const _Float16*)&o0;
    const _Float16* oh1 = (const _Float16*)&o1;

    #pragma unroll
    for (int t = 0; t < 16; ++t) {
        int   h = hc + t;
        float gv = (t < 8) ? (float)oh0[t] : (float)oh1[t - 8];
        float v  = ((acc[t] - mean) * rs * onscale[h] + onbias[h]) * gv;
        vt[jl * HID_ + h] = (f16)v;
    }
    __syncthreads();

    // Output projection: 8 waves -> (2 M-tiles) x (4 N-tiles), K=128 in 4 steps
    const int lane = tid & 31;
    const int wv   = tid >> 5;
    const int mt   = wv >> 2;
    const int nt   = wv & 3;
    const int lr   = lane & 15;
    const int g    = lane >> 4;
    const int ncol = nt * 16 + lr;

    v8f c;
    {
        float bias = bo[ncol];
        #pragma unroll
        for (int r = 0; r < 8; ++r) c[r] = bias;
    }
    const _Float16* arow = &vt[(mt * 16 + lr) * HID_];
    const f16*      brow = WoT + (long)ncol * HID_;
    #pragma unroll
    for (int kk = 0; kk < 4; ++kk) {
        int ko = kk * 32 + 8 * g;
        v16h a, b;
        ((uint4*)&a)[0] = *(const uint4*)(arow + ko);
        ((uint4*)&a)[1] = *(const uint4*)(arow + ko + 16);
        ((uint4*)&b)[0] = *(const uint4*)(brow + ko);
        ((uint4*)&b)[1] = *(const uint4*)(brow + ko + 16);
        c = __builtin_amdgcn_wmma_f32_16x16x32_f16(false, a, false, b, (short)0, c, false, false);
    }
    #pragma unroll
    for (int r = 0; r < 8; ++r) {
        long orow = (long)i * LL + (long)(jt * 32 + mt * 16 + r + 8 * g);
        out[orow * DIM_ + ncol] = c[r];
    }
}

// ---------------------------------------------------------------------------
extern "C" void kernel_launch(void* const* d_in, const int* in_sizes, int n_in,
                              void* d_out, int out_size, void* d_ws, size_t ws_size,
                              hipStream_t stream) {
    const float* x       = (const float*)d_in[0];
    const int*   smask   = (const int*)d_in[1];
    const float* nscale  = (const float*)d_in[2];
    const float* nbias   = (const float*)d_in[3];
    const float* Wl      = (const float*)d_in[4];
    const float* bl      = (const float*)d_in[5];
    const float* Wr      = (const float*)d_in[6];
    const float* br      = (const float*)d_in[7];
    const float* Wlg     = (const float*)d_in[8];
    const float* blg     = (const float*)d_in[9];
    const float* Wrg     = (const float*)d_in[10];
    const float* brg     = (const float*)d_in[11];
    const float* Wog     = (const float*)d_in[12];
    const float* bog     = (const float*)d_in[13];
    const float* onscale = (const float*)d_in[14];
    const float* onbias  = (const float*)d_in[15];
    const float* Wo      = (const float*)d_in[16];
    const float* bo      = (const float*)d_in[17];
    float*       out     = (float*)d_out;

    // Workspace layout (f16 intermediates + transposed weights)
    char* ws = (char*)d_ws;
    f16*  leftg  = (f16*)ws;                                   // 64 MB
    f16*  rightg = (f16*)(ws + (size_t)NROWS * HID_ * 2);      // 64 MB
    f16*  ogbuf  = (f16*)(ws + (size_t)NROWS * HID_ * 4);      // 64 MB
    f16*  W5T    = (f16*)(ws + (size_t)NROWS * HID_ * 6);      // 80 KB
    f16*  WoT    = W5T + 640 * 64;                             // 16 KB
    float* b5    = (float*)(WoT + 64 * 128);                   // 2.5 KB

    tm_prep_kernel<<<(640 * 64 + 255) / 256, 256, 0, stream>>>(
        Wl, Wr, Wlg, Wrg, Wog, Wo, bl, br, blg, brg, bog, W5T, WoT, b5);

    tm_proj_kernel<<<(int)(NROWS / 16), 256, 0, stream>>>(
        x, smask, nscale, nbias, W5T, b5, leftg, rightg, ogbuf);

    tm_band_out_kernel<<<LL * 16, 256, 0, stream>>>(
        leftg, rightg, ogbuf, onscale, onbias, WoT, bo, out);
}
